// MultiHeadAttention_4750233829530
// MI455X (gfx1250) — matrix operand order repaired and checked by reading, not compiled, not run on hardware
//
#include <hip/hip_runtime.h>
#include <hip/hip_bf16.h>

// ---------------------------------------------------------------------------
// MHA forward for MI455X (gfx1250, wave32, WMMA f32_16x16x32_f16).
// d_out = [ out: B*S*512 f32 | attn: B*H*S*S f32 ]
// ---------------------------------------------------------------------------

typedef __attribute__((ext_vector_type(16))) _Float16 v16h;
typedef __attribute__((ext_vector_type(8)))  _Float16 v8h;
typedef __attribute__((ext_vector_type(8)))  float    v8f;
typedef __attribute__((ext_vector_type(4)))  float    v4f;

constexpr int   D_MODEL = 512;
constexpr int   NH      = 8;
constexpr int   BATCH   = 4;
constexpr int   SEQ     = 2048;
constexpr int   TOK     = BATCH * SEQ;              // 8192
constexpr float SCALE   = 0.125f;                   // 1/sqrt(64)
constexpr float NEGV    = -1000000000.0f;
constexpr float LN_EPS  = 1e-5f;

// ---- WMMA fragment loaders ------------------------------------------------
// A 16x32 f16 layout: lanes 0-15 row M=lane, halves0-7=K0..7, halves8-15=K16..23
//                     lanes 16-31 row M=lane-16, halves0-7=K8..15, 8-15=K24..31
// B 32x16 f16 layout: lane holds column N=lane&15; same K order as A
//                     (halves0-7=K8h..8h+7, 8-15=K16+8h..).  (B supplied transposed: Bt[N][K])

__device__ __forceinline__ v16h frag_a_f16(const _Float16* A, int lda) {
  const int lane = threadIdx.x & 31;
  const _Float16* p = A + (lane & 15) * (size_t)lda + ((lane >> 4) << 3);
  v8h lo = *(const v8h*)(p);
  v8h hi = *(const v8h*)(p + 16);
  return __builtin_shufflevector(lo, hi, 0,1,2,3,4,5,6,7,8,9,10,11,12,13,14,15);
}

__device__ __forceinline__ v16h frag_a_f32(const float* A, int lda) {
  const int lane = threadIdx.x & 31;
  const float* p = A + (lane & 15) * (size_t)lda + ((lane >> 4) << 3);
  v4f f0 = *(const v4f*)(p);
  v4f f1 = *(const v4f*)(p + 4);
  v4f f2 = *(const v4f*)(p + 16);
  v4f f3 = *(const v4f*)(p + 20);
  v16h a;
#pragma unroll
  for (int i = 0; i < 4; ++i) {
    a[i]      = (_Float16)f0[i];
    a[4 + i]  = (_Float16)f1[i];
    a[8 + i]  = (_Float16)f2[i];
    a[12 + i] = (_Float16)f3[i];
  }
  return a;
}

__device__ __forceinline__ v16h frag_b_t(const _Float16* Bt, int ldb) {
  const int lane = threadIdx.x & 31;
  const _Float16* p = Bt + (lane & 15) * (size_t)ldb + ((lane >> 4) << 3);
  v8h lo = *(const v8h*)(p);
  v8h hi = *(const v8h*)(p + 16);
  return __builtin_shufflevector(lo, hi, 0,1,2,3,4,5,6,7,8,9,10,11,12,13,14,15);
}

__device__ __forceinline__ v8f wmma_f16(v16h a, v16h b, v8f c) {
  return __builtin_amdgcn_wmma_f32_16x16x32_f16(false, a, false, b, (short)0, c,
                                                false, false);
}

// ---- 1. transpose + f16-convert the four 512x512 weight matrices -----------
__global__ __launch_bounds__(256) void k_prep_weights(
    const float* __restrict__ Wq, const float* __restrict__ Wk,
    const float* __restrict__ Wv, const float* __restrict__ Wf,
    _Float16* __restrict__ o) {
  int idx = blockIdx.x * 256 + threadIdx.x;           // 4*512*512 total
  if (idx >= 4 * 512 * 512) return;
  int w = idx >> 18;
  int n = (idx >> 9) & 511;
  int k = idx & 511;
  const float* src = (w == 0) ? Wq : (w == 1) ? Wk : (w == 2) ? Wv : Wf;
  o[idx] = (_Float16)src[k * 512 + n];                // Wt[n][k] = W[k][n]
}

// ---- 2. QKV projection: [TOK,512] x Wt -> Q/K f16 [b,h,S,64], V f16 [b,h,64,S]
__global__ __launch_bounds__(32) void k_proj_qkv(
    const float* __restrict__ inQ, const float* __restrict__ inK,
    const float* __restrict__ inV, const _Float16* __restrict__ Wt,
    _Float16* __restrict__ Qh, _Float16* __restrict__ Kh,
    _Float16* __restrict__ Vt) {
  const int which = blockIdx.z;
  const int m0 = blockIdx.y * 16;
  const int n0 = blockIdx.x * 16;
  const float* A = (which == 0) ? inQ : (which == 1) ? inK : inV;
  const _Float16* Bt = Wt + (size_t)which * 262144 + (size_t)n0 * 512;
  v8f acc = {};
#pragma unroll 4
  for (int k = 0; k < 512; k += 32) {
    v16h a = frag_a_f32(A + (size_t)m0 * 512 + k, 512);
    v16h b = frag_b_t(Bt + k, 512);
    acc = wmma_f16(a, b, acc);
  }
  const int lane = threadIdx.x & 31;
  const int col = lane & 15, rb = (lane >> 4) << 3;
#pragma unroll
  for (int j = 0; j < 8; ++j) {
    int tok = m0 + rb + j;
    int n = n0 + col;
    int bb = tok >> 11, s = tok & 2047;
    int hh = n >> 6,   d = n & 63;
    size_t head = (size_t)(bb * NH + hh);
    _Float16 v = (_Float16)acc[j];
    if (which == 0)      Qh[(head * SEQ + s) * 64 + d] = v;
    else if (which == 1) Kh[(head * SEQ + s) * 64 + d] = v;
    else                 Vt[(head * 64 + d) * SEQ + s] = v;
  }
}

// ---- 3. scores = scale * Q Kt, masked -> attn (f32, in d_out) ---------------
__global__ __launch_bounds__(32) void k_scores(
    const _Float16* __restrict__ Qh, const _Float16* __restrict__ Kh,
    const unsigned char* __restrict__ mask, float* __restrict__ attn) {
  const int bh = blockIdx.z;
  const int q0 = blockIdx.y * 16;
  const int c0 = blockIdx.x * 16;
  const _Float16* Qb = Qh + (size_t)bh * SEQ * 64 + (size_t)q0 * 64;
  const _Float16* Kb = Kh + (size_t)bh * SEQ * 64 + (size_t)c0 * 64;
  v8f acc = {};
#pragma unroll
  for (int k = 0; k < 64; k += 32) {
    v16h a = frag_a_f16(Qb + k, 64);
    v16h b = frag_b_t(Kb + k, 64);
    acc = wmma_f16(a, b, acc);
  }
  const int lane = threadIdx.x & 31;
  const int col = lane & 15, rb = (lane >> 4) << 3;
  const int bb = bh >> 3;
  const unsigned char* mrow = mask + (size_t)bb * SEQ * SEQ;
  float* orow = attn + (size_t)bh * SEQ * SEQ;
#pragma unroll
  for (int j = 0; j < 8; ++j) {
    size_t q = q0 + rb + j;
    size_t c = c0 + col;
    float v = acc[j] * SCALE;
    v = mrow[q * SEQ + c] ? NEGV : v;
    orow[q * SEQ + c] = v;
  }
}

// ---- 4. row softmax over 2048 keys, in place in d_out ----------------------
__global__ __launch_bounds__(256) void k_softmax(float* __restrict__ attn) {
  __shared__ float red[256];
  const int t = threadIdx.x;
  float* row = attn + (size_t)blockIdx.x * SEQ;
  float v[8];
  float m = -3.4e38f;
#pragma unroll
  for (int i = 0; i < 8; ++i) { v[i] = row[t + i * 256]; m = fmaxf(m, v[i]); }
  red[t] = m; __syncthreads();
  for (int o = 128; o > 0; o >>= 1) {
    if (t < o) red[t] = fmaxf(red[t], red[t + o]);
    __syncthreads();
  }
  m = red[0]; __syncthreads();
  float s = 0.f;
#pragma unroll
  for (int i = 0; i < 8; ++i) { v[i] = __expf(v[i] - m); s += v[i]; }
  red[t] = s; __syncthreads();
  for (int o = 128; o > 0; o >>= 1) {
    if (t < o) red[t] += red[t + o];
    __syncthreads();
  }
  float inv = 1.0f / red[0];
#pragma unroll
  for (int i = 0; i < 8; ++i) row[t + i * 256] = v[i] * inv;
}

// ---- 5. context = P @ V  -> ctx f16 [TOK, 512] -----------------------------
// One workgroup computes the FULL 16x64 output row-block of a head:
// each P (attn) fragment is read from HBM exactly once and multiplied
// against 4 V tiles (4 accumulators) -> 4x less traffic on the dominant
// 536 MB attn read stream, and 4 independent WMMAs per A-load for ILP.
__global__ __launch_bounds__(32) void k_context(
    const float* __restrict__ attn, const _Float16* __restrict__ Vt,
    _Float16* __restrict__ ctx) {
  const int bh = blockIdx.y;
  const int q0 = blockIdx.x * 16;
  const float* Pb = attn + (size_t)bh * SEQ * SEQ + (size_t)q0 * SEQ;
  const _Float16* Vb = Vt + (size_t)bh * 64 * SEQ;
  v8f acc0 = {}, acc1 = {}, acc2 = {}, acc3 = {};
#pragma unroll 2
  for (int k = 0; k < SEQ; k += 32) {
    v16h a  = frag_a_f32(Pb + k, SEQ);
    v16h b0 = frag_b_t(Vb + k, SEQ);
    v16h b1 = frag_b_t(Vb + (size_t)16 * SEQ + k, SEQ);
    v16h b2 = frag_b_t(Vb + (size_t)32 * SEQ + k, SEQ);
    v16h b3 = frag_b_t(Vb + (size_t)48 * SEQ + k, SEQ);
    acc0 = wmma_f16(a, b0, acc0);
    acc1 = wmma_f16(a, b1, acc1);
    acc2 = wmma_f16(a, b2, acc2);
    acc3 = wmma_f16(a, b3, acc3);
  }
  const int lane = threadIdx.x & 31;
  const int col = lane & 15, rb = (lane >> 4) << 3;
  const int bb = bh >> 3, hh = bh & 7;
#pragma unroll
  for (int j = 0; j < 8; ++j) {
    size_t tok = (size_t)bb * SEQ + q0 + rb + j;
    _Float16* crow = ctx + tok * 512 + hh * 64 + col;
    crow[0]  = (_Float16)acc0[j];
    crow[16] = (_Float16)acc1[j];
    crow[32] = (_Float16)acc2[j];
    crow[48] = (_Float16)acc3[j];
  }
}

// ---- 6. fcout = ctx @ W_fc  (f32) ------------------------------------------
__global__ __launch_bounds__(32) void k_fc(
    const _Float16* __restrict__ ctx, const _Float16* __restrict__ WtF,
    float* __restrict__ fcout) {
  const int m0 = blockIdx.y * 16;
  const int n0 = blockIdx.x * 16;
  v8f acc = {};
#pragma unroll 4
  for (int k = 0; k < 512; k += 32) {
    v16h a = frag_a_f16(ctx + (size_t)m0 * 512 + k, 512);
    v16h b = frag_b_t(WtF + (size_t)n0 * 512 + k, 512);
    acc = wmma_f16(a, b, acc);
  }
  const int lane = threadIdx.x & 31;
  const int col = lane & 15, rb = (lane >> 4) << 3;
#pragma unroll
  for (int j = 0; j < 8; ++j)
    fcout[(size_t)(m0 + rb + j) * 512 + n0 + col] = acc[j];
}

// ---- 7. residual + LayerNorm -> out (d_out region 0) -----------------------
__global__ __launch_bounds__(256) void k_ln(
    const float* __restrict__ fcout, const float* __restrict__ inQ,
    const float* __restrict__ gamma, const float* __restrict__ beta,
    float* __restrict__ out) {
  __shared__ float rs[256];
  __shared__ float rq[256];
  const int t = threadIdx.x;
  const size_t base = (size_t)blockIdx.x * 512;
  float y0 = fcout[base + t]       + inQ[base + t];
  float y1 = fcout[base + 256 + t] + inQ[base + 256 + t];
  rs[t] = y0 + y1;
  rq[t] = y0 * y0 + y1 * y1;
  __syncthreads();
  for (int o = 128; o > 0; o >>= 1) {
    if (t < o) { rs[t] += rs[t + o]; rq[t] += rq[t + o]; }
    __syncthreads();
  }
  float mu  = rs[0] * (1.0f / 512.0f);
  float var = rq[0] * (1.0f / 512.0f) - mu * mu;
  float inv = rsqrtf(var + LN_EPS);
  out[base + t]       = (y0 - mu) * inv * gamma[t]       + beta[t];
  out[base + 256 + t] = (y1 - mu) * inv * gamma[t + 256] + beta[t + 256];
}

// ---------------------------------------------------------------------------
extern "C" void kernel_launch(void* const* d_in, const int* in_sizes, int n_in,
                              void* d_out, int out_size, void* d_ws,
                              size_t ws_size, hipStream_t stream) {
  (void)in_sizes; (void)n_in; (void)out_size; (void)ws_size;
  const float* inQ   = (const float*)d_in[0];
  const float* inK   = (const float*)d_in[1];
  const float* inV   = (const float*)d_in[2];
  const unsigned char* mask = (const unsigned char*)d_in[3];  // numpy bool = 1B
  const float* Wq    = (const float*)d_in[4];
  const float* Wk    = (const float*)d_in[5];
  const float* Wv    = (const float*)d_in[6];
  const float* Wf    = (const float*)d_in[7];
  const float* gamma = (const float*)d_in[8];
  const float* beta  = (const float*)d_in[9];

  float* out  = (float*)d_out;
  float* attn = out + (size_t)BATCH * SEQ * D_MODEL;

  // workspace: 4 f16 weight mats | Q,K f16 [b,h,S,64] | V f16 [b,h,64,S]
  //            | ctx f16 [TOK,512] | fcout f32 [TOK,512]   (~52 MB total)
  _Float16* Wt  = (_Float16*)d_ws;
  _Float16* Qh  = Wt + 4 * 262144;
  _Float16* Kh  = Qh + (size_t)TOK * 512;
  _Float16* Vt  = Kh + (size_t)TOK * 512;
  _Float16* Ctx = Vt + (size_t)TOK * 512;
  float*    fco = (float*)(Ctx + (size_t)TOK * 512);

  k_prep_weights<<<4096, 256, 0, stream>>>(Wq, Wk, Wv, Wf, Wt);
  k_proj_qkv<<<dim3(D_MODEL / 16, TOK / 16, 3), 32, 0, stream>>>(
      inQ, inK, inV, Wt, Qh, Kh, Vt);
  k_scores<<<dim3(SEQ / 16, SEQ / 16, BATCH * NH), 32, 0, stream>>>(
      Qh, Kh, mask, attn);
  k_softmax<<<BATCH * NH * SEQ, 256, 0, stream>>>(attn);
  k_context<<<dim3(SEQ / 16, BATCH * NH), 32, 0, stream>>>(attn, Vt, Ctx);
  k_fc<<<dim3(D_MODEL / 16, TOK / 16), 32, 0, stream>>>(Ctx, Wt + 3 * 262144, fco);
  k_ln<<<TOK, 256, 0, stream>>>(fco, inQ, gamma, beta, out);
}
